// GATEncoder_35124242546939
// MI455X (gfx1250) — matrix operand order repaired and checked by reading, not compiled, not run on hardware
//
#include <hip/hip_runtime.h>
#include <hip/hip_bf16.h>
#include <stdint.h>

// ---------------------------------------------------------------------------
// Problem constants (match reference)
// ---------------------------------------------------------------------------
#define NNODES 50000
#define NEDGES 800000
#define IN_F   128
#define HIDD   64
#define NEG_SLOPE 0.2f
#define MPAD   50048          // N padded to multiple of 128 for GEMM tiling

typedef __attribute__((ext_vector_type(16))) __bf16 v16bf;
typedef __attribute__((ext_vector_type(8)))  float  v8f;

struct alignas(16) qword { unsigned u[4]; };
union frag32 { qword q[2]; v16bf v; };     // 32 bytes = one WMMA bf16 fragment

// ---------------------------------------------------------------------------
// Helpers
// ---------------------------------------------------------------------------
__device__ __forceinline__ unsigned enc_f32(float f) {
    unsigned u = __float_as_uint(f);
    return (u & 0x80000000u) ? ~u : (u | 0x80000000u);   // monotone float->uint
}
__device__ __forceinline__ float dec_f32(unsigned k) {
    unsigned u = (k & 0x80000000u) ? (k & 0x7FFFFFFFu) : ~k;
    return __uint_as_float(u);
}
#define MKEY_NEG_INF 0x007FFFFFu   // enc_f32(-inf)

// ---------------------------------------------------------------------------
// Fill kernels
// ---------------------------------------------------------------------------
__global__ void fill_f32_kernel(float* __restrict__ p, float v, int n) {
    int i = blockIdx.x * blockDim.x + threadIdx.x;
    if (i < n) p[i] = v;
}
__global__ void fill_u32_kernel(unsigned* __restrict__ p, unsigned v, int n) {
    int i = blockIdx.x * blockDim.x + threadIdx.x;
    if (i < n) p[i] = v;
}

// ---------------------------------------------------------------------------
// fp32 -> bf16 staging.  XB is [MPAD x K] row-major, zero-padded rows.
// ---------------------------------------------------------------------------
__global__ void pack_x_kernel(const float* __restrict__ x, __bf16* __restrict__ xb,
                              int M, int K) {
    int i = blockIdx.x * blockDim.x + threadIdx.x;
    int total = MPAD * K;
    if (i >= total) return;
    int row = i / K;
    xb[i] = (row < M) ? (__bf16)x[i] : (__bf16)0.0f;
}

// W [K x C] row-major fp32  ->  Bt [C x K] row-major bf16 (column-major W)
__global__ void pack_w_kernel(const float* __restrict__ W, __bf16* __restrict__ Bt,
                              int K, int C) {
    int i = blockIdx.x * blockDim.x + threadIdx.x;
    if (i >= K * C) return;
    int k = i / C, n = i % C;
    Bt[(size_t)n * K + k] = (__bf16)W[i];
}

// ---------------------------------------------------------------------------
// WMMA GEMM:  Cmat[M x C] = A[MPAD x K] * B[K x C]   (B given transposed: Bt[C x K])
// Block = 256 threads = 8 waves.  Wave w -> rows [by*128 + w*16, +16),
// cols [bx*64, +64) as 4 accumulators of v_wmma_f32_16x16x32_bf16.
//
// The 64-column B strip for the block (<= 64 x 256 bf16 = 32KB) is staged in
// LDS once, with a 16B per-row pad so the 16 column-readers hit distinct
// bank groups (row stride = K+8 bf16 -> 132 dwords -> bank stride 4 mod 64).
// ---------------------------------------------------------------------------
__global__ __launch_bounds__(256)
void gemm_wmma_kernel(const __bf16* __restrict__ A, const __bf16* __restrict__ Bt,
                      float* __restrict__ Cmat, int M, int K, int C) {
    __shared__ __bf16 Bs[64 * (256 + 8)];          // max K = 256, padded rows

    const int lane = threadIdx.x & 31;
    const int wave = threadIdx.x >> 5;
    const int tileM = blockIdx.y * 128 + wave * 16;
    const int tileN = blockIdx.x * 64;
    const int m  = lane & 15;
    const int hs = lane >> 4;
    const int Kp = K + 8;                          // padded LDS row stride (bf16)

    // ---- stage B strip: rows [tileN, tileN+64) of Bt, contiguous in global ----
    {
        const int KV = K >> 3;                     // 16B vectors per row
        const qword* gsrc = (const qword*)(Bt + (size_t)tileN * K);
        qword* ldst = (qword*)Bs;
        for (int i = threadIdx.x; i < 64 * KV; i += 256) {
            int r = i / KV, c = i - r * KV;
            ldst[(size_t)r * (KV + 1) + c] = gsrc[i];
        }
    }
    __syncthreads();

    v8f acc0 = {}, acc1 = {}, acc2 = {}, acc3 = {};
    const __bf16* Arow = A + (size_t)(tileM + m) * K;

    for (int k0 = 0; k0 < K; k0 += 32) {
        // A fragment (16x32): elems 0..7 <- K[k0+hs*8 ..), 8..15 <- K[k0+16+hs*8 ..)
        frag32 a;
        a.q[0] = *(const qword*)(Arow + k0 + hs * 8);
        a.q[1] = *(const qword*)(Arow + k0 + 16 + hs * 8);
        // 4 B tiles (32x16 each) out of LDS; A reused across them
#pragma unroll
        for (int t = 0; t < 4; ++t) {
            const __bf16* Brow = Bs + (size_t)(t * 16 + m) * Kp + k0 + hs * 8;
            frag32 b;
            b.q[0] = *(const qword*)(Brow);
            b.q[1] = *(const qword*)(Brow + 16);
            v8f* acc = (t == 0) ? &acc0 : (t == 1) ? &acc1 : (t == 2) ? &acc2 : &acc3;
            *acc = __builtin_amdgcn_wmma_f32_16x16x32_bf16(
                false, a.v, false, b.v, (short)0, *acc, false, false);
        }
    }

    // ---- Store: C/D layout — VGPR r, lanes0-15: (M=r, N=lane); lanes16-31: (M=r+8)
#pragma unroll
    for (int t = 0; t < 4; ++t) {
        v8f acc = (t == 0) ? acc0 : (t == 1) ? acc1 : (t == 2) ? acc2 : acc3;
#pragma unroll
        for (int r = 0; r < 8; ++r) {
            int row = tileM + r + hs * 8;
            if (row < M)
                Cmat[(size_t)row * C + tileN + t * 16 + m] = acc[r];
        }
    }
}

// ---------------------------------------------------------------------------
// Attention logits: el[n,h] = <hbuf[n,h,:], al[h,:]> ; er likewise.
// ---------------------------------------------------------------------------
__global__ void logits_kernel(const float* __restrict__ h,
                              const float* __restrict__ al,
                              const float* __restrict__ ar,
                              float* __restrict__ el, float* __restrict__ er,
                              int H) {
    int i = blockIdx.x * blockDim.x + threadIdx.x;
    if (i >= NNODES * H) return;
    int n = i / H, hh = i % H;
    const float* hp = h + (size_t)n * H * HIDD + hh * HIDD;
    const float* alp = al + hh * HIDD;
    const float* arp = ar + hh * HIDD;
    float sl = 0.f, sr = 0.f;
#pragma unroll 8
    for (int d = 0; d < HIDD; ++d) {
        float v = hp[d];
        sl += v * alp[d];
        sr += v * arp[d];
    }
    el[i] = sl;
    er[i] = sr;
}

// ---------------------------------------------------------------------------
// Pass 1: e = leaky_relu(el[src]+er[dst]); segment max via atomicMax on keys.
// ---------------------------------------------------------------------------
__global__ void edge_max_kernel(const int* __restrict__ src, const int* __restrict__ dst,
                                const float* __restrict__ el, const float* __restrict__ er,
                                unsigned* __restrict__ mkey, int H) {
    int i = blockIdx.x * blockDim.x + threadIdx.x;
    if (i >= NEDGES * H) return;
    int e  = i / H, hh = i % H;
    int s  = src[e], d = dst[e];
    float v = el[s * H + hh] + er[d * H + hh];
    v = (v >= 0.f) ? v : NEG_SLOPE * v;
    atomicMax(&mkey[d * H + hh], enc_f32(v));
}

__global__ void m_fix_kernel(const unsigned* __restrict__ mkey,
                             float* __restrict__ mf, int n) {
    int i = blockIdx.x * blockDim.x + threadIdx.x;
    if (i >= n) return;
    unsigned k = mkey[i];
    mf[i] = (k == MKEY_NEG_INF) ? 0.f : dec_f32(k);   // empty segments -> 0
}

// ---------------------------------------------------------------------------
// Pass 2: ex = exp(e - m[dst]);  s[dst] += ex;  store ex.
// ---------------------------------------------------------------------------
__global__ void edge_sum_kernel(const int* __restrict__ src, const int* __restrict__ dst,
                                const float* __restrict__ el, const float* __restrict__ er,
                                const float* __restrict__ mf,
                                float* __restrict__ exbuf, float* __restrict__ ssum,
                                int H) {
    int i = blockIdx.x * blockDim.x + threadIdx.x;
    if (i >= NEDGES * H) return;
    int e  = i / H, hh = i % H;
    int s  = src[e], d = dst[e];
    float v = el[s * H + hh] + er[d * H + hh];
    v = (v >= 0.f) ? v : NEG_SLOPE * v;
    float ex = __expf(v - mf[d * H + hh]);
    exbuf[i] = ex;
    atomicAdd(&ssum[d * H + hh], ex);
}

// ---------------------------------------------------------------------------
// Pass 3: out[dst] += h[src] * alpha.   One wave32 per edge; lanes stride
// the C channels -> coalesced loads + coalesced global_atomic_add_f32.
// ---------------------------------------------------------------------------
__global__ __launch_bounds__(256)
void scatter_kernel(const int* __restrict__ src, const int* __restrict__ dst,
                    const float* __restrict__ h, const float* __restrict__ exbuf,
                    const float* __restrict__ ssum, float* __restrict__ out,
                    int H, int C) {
    int wid  = (blockIdx.x * blockDim.x + threadIdx.x) >> 5;
    int lane = threadIdx.x & 31;
    if (wid >= NEDGES) return;
    int s = src[wid], d = dst[wid];
    const float* hp = h + (size_t)s * C;
    float* op = out + (size_t)d * C;
    for (int c = lane; c < C; c += 32) {
        int head = c >> 6;   // HIDD == 64
        float alpha = exbuf[wid * H + head] / ssum[d * H + head];
        atomicAdd(&op[c], hp[c] * alpha);
    }
}

// ---------------------------------------------------------------------------
// Finalize: act = elu(out + b)   /   d_out = out + b (last layer, head mean
// over a single head is identity).
// ---------------------------------------------------------------------------
__global__ void bias_elu_kernel(const float* __restrict__ out, const float* __restrict__ b,
                                float* __restrict__ act, int C) {
    int i = blockIdx.x * blockDim.x + threadIdx.x;
    if (i >= NNODES * C) return;
    float v = out[i] + b[i % C];
    act[i] = (v > 0.f) ? v : (__expf(v) - 1.f);
}
__global__ void bias_kernel(const float* __restrict__ out, const float* __restrict__ b,
                            float* __restrict__ dst, int C) {
    int i = blockIdx.x * blockDim.x + threadIdx.x;
    if (i >= NNODES * C) return;
    dst[i] = out[i] + b[i % C];
}

// ---------------------------------------------------------------------------
// Host-side orchestration
// ---------------------------------------------------------------------------
static inline int cdiv(long long a, int b) { return (int)((a + b - 1) / b); }

struct Ws {
    __bf16*   XB;   // [MPAD x 256] bf16 staging of activations
    __bf16*   WB;   // [256 x 256] bf16 transposed weights
    float*    H;    // [N x 256] gemm output
    float*    OUT;  // [N x 256] aggregation accumulator
    float*    ACT;  // [N x 256] layer activation
    float*    EL;   // [N x 4]
    float*    ER;   // [N x 4]
    unsigned* MK;   // [N x 4] max keys
    float*    MF;   // [N x 4] max values
    float*    S;    // [N x 4] softmax denominators
    float*    EX;   // [E x 4] exp(e - m)
};

static void run_layer(const float* x, int K, int H, int C,
                      const float* W, const float* al, const float* ar, const float* b,
                      const int* src, const int* dst,
                      const Ws& w, float* final_dst, bool last, hipStream_t st) {
    const int T = 256;
    // stage inputs
    pack_x_kernel<<<cdiv((long long)MPAD * K, T), T, 0, st>>>(x, w.XB, NNODES, K);
    pack_w_kernel<<<cdiv((long long)K * C, T), T, 0, st>>>(W, w.WB, K, C);
    // dense GEMM on the matrix pipes
    dim3 gg(C / 64, MPAD / 128);
    gemm_wmma_kernel<<<gg, T, 0, st>>>(w.XB, w.WB, w.H, NNODES, K, C);
    // attention logits
    logits_kernel<<<cdiv((long long)NNODES * H, T), T, 0, st>>>(w.H, al, ar, w.EL, w.ER, H);
    // init accumulators
    fill_u32_kernel<<<cdiv((long long)NNODES * H, T), T, 0, st>>>(w.MK, MKEY_NEG_INF, NNODES * H);
    fill_f32_kernel<<<cdiv((long long)NNODES * H, T), T, 0, st>>>(w.S, 0.f, NNODES * H);
    fill_f32_kernel<<<cdiv((long long)NNODES * C, T), T, 0, st>>>(w.OUT, 0.f, NNODES * C);
    // segment softmax over edges
    edge_max_kernel<<<cdiv((long long)NEDGES * H, T), T, 0, st>>>(src, dst, w.EL, w.ER, w.MK, H);
    m_fix_kernel<<<cdiv((long long)NNODES * H, T), T, 0, st>>>(w.MK, w.MF, NNODES * H);
    edge_sum_kernel<<<cdiv((long long)NEDGES * H, T), T, 0, st>>>(src, dst, w.EL, w.ER, w.MF,
                                                                  w.EX, w.S, H);
    // weighted scatter-sum (wave per edge)
    scatter_kernel<<<cdiv((long long)NEDGES * 32, T), T, 0, st>>>(src, dst, w.H, w.EX, w.S,
                                                                  w.OUT, H, C);
    // bias (+ activation)
    if (last)
        bias_kernel<<<cdiv((long long)NNODES * C, T), T, 0, st>>>(w.OUT, b, final_dst, C);
    else
        bias_elu_kernel<<<cdiv((long long)NNODES * C, T), T, 0, st>>>(w.OUT, b, w.ACT, C);
}

extern "C" void kernel_launch(void* const* d_in, const int* in_sizes, int n_in,
                              void* d_out, int out_size, void* d_ws, size_t ws_size,
                              hipStream_t stream) {
    const float* feat = (const float*)d_in[0];
    const int*   src  = (const int*)d_in[1];
    const int*   dst  = (const int*)d_in[2];
    const float* W1 = (const float*)d_in[3],  *al1 = (const float*)d_in[4],
               * ar1 = (const float*)d_in[5], *b1  = (const float*)d_in[6];
    const float* W2 = (const float*)d_in[7],  *al2 = (const float*)d_in[8],
               * ar2 = (const float*)d_in[9], *b2  = (const float*)d_in[10];
    const float* W3 = (const float*)d_in[11], *al3 = (const float*)d_in[12],
               * ar3 = (const float*)d_in[13], *b3 = (const float*)d_in[14];

    // carve workspace (256B aligned slices)
    char* base = (char*)d_ws;
    size_t off = 0;
    auto carve = [&](size_t bytes) {
        void* p = base + off;
        off += (bytes + 255) & ~(size_t)255;
        return p;
    };
    Ws w;
    w.XB  = (__bf16*)  carve((size_t)MPAD * 256 * 2);
    w.WB  = (__bf16*)  carve((size_t)256 * 256 * 2);
    w.H   = (float*)   carve((size_t)NNODES * 256 * 4);
    w.OUT = (float*)   carve((size_t)NNODES * 256 * 4);
    w.ACT = (float*)   carve((size_t)NNODES * 256 * 4);
    w.EL  = (float*)   carve((size_t)NNODES * 4 * 4);
    w.ER  = (float*)   carve((size_t)NNODES * 4 * 4);
    w.MK  = (unsigned*)carve((size_t)NNODES * 4 * 4);
    w.MF  = (float*)   carve((size_t)NNODES * 4 * 4);
    w.S   = (float*)   carve((size_t)NNODES * 4 * 4);
    w.EX  = (float*)   carve((size_t)NEDGES * 4 * 4);
    (void)ws_size; (void)n_in; (void)in_sizes; (void)out_size;

    // Layer 1: IN=128 -> 4 heads x 64,  ELU
    run_layer(feat, IN_F, 4, 256, W1, al1, ar1, b1, src, dst, w, nullptr, false, stream);
    // Layer 2: 256 -> 4 heads x 64,  ELU
    run_layer(w.ACT, 256, 4, 256, W2, al2, ar2, b2, src, dst, w, nullptr, false, stream);
    // Layer 3: 256 -> 1 head x 64, no activation; head-mean is identity
    run_layer(w.ACT, 256, 1, 64, W3, al3, ar3, b3, src, dst, w, (float*)d_out, true, stream);
}
